// InteractionModule_14663018348662
// MI455X (gfx1250) — compile-verified, matrix operand order checked by reading, not run on hardware
//
#include <hip/hip_runtime.h>
#include <math.h>

#define EPSV 1e-6f
#define INV_SQRT3 0.57735026918962576451f
#define INV_SQRT2 0.70710678118654752440f

typedef __attribute__((ext_vector_type(16))) _Float16 v16h;
typedef __attribute__((ext_vector_type(8)))  float    v8f;

union AF { v16h v; unsigned u[8]; _Float16 h[16]; };

// fast hardware transcendentals (v_rcp_f32 / v_rsq_f32)
__device__ inline float frcp(float x) { return __builtin_amdgcn_rcpf(x); }
__device__ inline float frsq(float x) { return __builtin_amdgcn_rsqf(x); }

// A-fragment (16x32 f16) from an LDS row-major f16 panel.
// ISA layout: lanes 0-15 -> M=lane, lanes 16-31 -> M=lane-16; per lane the 16
// halves cover K = {8g+0..7, 16+8g+0..7} (g = lane>>4), packed in consecutive pairs.
__device__ inline v16h a_frag_lds(const _Float16* panel, int stride_h, int kbase) {
  const int l = threadIdx.x & 31;
  const int r = l & 15;
  const int g = l >> 4;
  AF f;
  const _Float16* rowp = panel + r * stride_h + kbase;
#pragma unroll
  for (int i = 0; i < 8; ++i) {
    const int k = (i < 4) ? (8 * g + 2 * i) : (16 + 8 * g + 2 * (i - 4));
    f.u[i] = *(const unsigned*)(rowp + k);
  }
  return f.v;
}

// B-fragment from prepacked weights: tile = 32 lanes x 16 halves contiguous.
__device__ inline v16h b_frag(const _Float16* pack, int tile) {
  return *(const v16h*)(pack + (size_t)tile * 512 + (threadIdx.x & 31) * 16);
}

__device__ inline v8f wmma_f16(v16h a, v16h b, v8f c) {
  return __builtin_amdgcn_wmma_f32_16x16x32_f16(false, a, false, b, (short)0, c,
                                                false, false);
}

__device__ inline float silu_f(float x) { return x * frcp(1.0f + __expf(-x)); }

__device__ inline unsigned fenc(float f) {
  unsigned u = __float_as_uint(f);
  return (u & 0x80000000u) ? ~u : (u | 0x80000000u);
}
__device__ inline float fdec(unsigned e) {
  return (e & 0x80000000u) ? __uint_as_float(e & 0x7fffffffu) : __uint_as_float(~e);
}

// butterfly sum across the 16-lane half-wave (lanes stay within their half)
__device__ inline float hsum16(float p) {
#pragma unroll
  for (int off = 1; off < 16; off <<= 1) p += __shfl_xor(p, off, 32);
  return p;
}

// ---------------------------------------------------------------------------
// Weight prepack: f32 [K,N] row-major -> f16 B-fragment tiles.
// Tile bases (in tiles): env1=0(40) env2=40(8) wq=48(32) wk=80(64) p1=144(16) p2=160(40)
// ---------------------------------------------------------------------------
__global__ void pack_weights(const float* __restrict__ env1, const float* __restrict__ env2,
                             const float* __restrict__ wq,   const float* __restrict__ wk,
                             const float* __restrict__ p1,   const float* __restrict__ p2,
                             _Float16* __restrict__ pack) {
  const int b = blockIdx.x;
  const float* src; int Nn; int bt; int tbase;
  if (b < 40)       { src = env1; Nn = 128; bt = b;       tbase = 0;   }
  else if (b < 48)  { src = env2; Nn = 32;  bt = b - 40;  tbase = 40;  }
  else if (b < 80)  { src = wq;   Nn = 256; bt = b - 48;  tbase = 48;  }
  else if (b < 144) { src = wk;   Nn = 256; bt = b - 80;  tbase = 80;  }
  else if (b < 160) { src = p1;   Nn = 128; bt = b - 144; tbase = 144; }
  else              { src = p2;   Nn = 160; bt = b - 160; tbase = 160; }
  const int Nt = Nn >> 4;
  const int kt = bt / Nt, nt = bt % Nt;
  const int l = threadIdx.x;
  const int g = l >> 4, lo = l & 15;
  _Float16* dst = pack + (size_t)(tbase + bt) * 512 + l * 16;
  const int n = nt * 16 + lo;
#pragma unroll
  for (int i = 0; i < 8; ++i) {
    const int k0 = kt * 32 + ((i < 4) ? (8 * g + 2 * i) : (16 + 8 * g + 2 * (i - 4)));
    dst[2 * i]     = (_Float16)src[(size_t)k0 * Nn + n];
    dst[2 * i + 1] = (_Float16)src[(size_t)(k0 + 1) * Nn + n];
  }
}

// ---------------------------------------------------------------------------
// K1: per 16-edge group (1 wave): h = silu([s,cond]W1+b1); w = hW2+b2;
//     Q = na[src]Wq; K = sWk; logits = clip(QK^T/4); atomicMax into mx.
// ---------------------------------------------------------------------------
__global__ void __launch_bounds__(128)
k1_edge_gemm(const float* __restrict__ scal, const float* __restrict__ cond,
             const float* __restrict__ nattr, const int* __restrict__ esrc,
             const float* __restrict__ b1, const float* __restrict__ b2,
             const _Float16* __restrict__ pack,
             float* __restrict__ logits, float* __restrict__ wenv,
             unsigned* __restrict__ mx, int E) {
  __shared__ _Float16 sh[4 * (16 * 160 + 16 * 64 + 16 * 128)];
  const int wave = threadIdx.x >> 5;
  const int l = threadIdx.x & 31;
  const int lo = l & 15, hi = l >> 4;
  _Float16* a1 = sh + wave * (16 * 160 + 16 * 64 + 16 * 128); // [16][160] s|cond
  _Float16* qp = a1 + 16 * 160;                               // [16][64]  gathered node_attrs
  _Float16* hp = qp + 16 * 64;                                // [16][128] h (post-silu)
  const int e0 = (blockIdx.x * 4 + wave) * 16;
  const bool act = (e0 < E);

  if (act) {
    for (int row = 0; row < 16; ++row) {
      const float* srow = scal + (size_t)(e0 + row) * 128;
      const float* crow = cond + (size_t)(e0 + row) * 32;
      for (int c = l; c < 160; c += 32)
        a1[row * 160 + c] = (_Float16)((c < 128) ? srow[c] : crow[c - 128]);
      const float* nrow = nattr + (size_t)esrc[e0 + row] * 64;
      qp[row * 64 + l]      = (_Float16)nrow[l];
      qp[row * 64 + l + 32] = (_Float16)nrow[l + 32];
    }
  }
  __syncthreads();

  if (act) { // GEMM1: [16,160]x[160,128] -> h
#pragma unroll 1
    for (int nt = 0; nt < 8; ++nt) {
      v8f acc = {};
#pragma unroll
      for (int kt = 0; kt < 5; ++kt)
        acc = wmma_f16(a_frag_lds(a1, 160, kt * 32), b_frag(pack, kt * 8 + nt), acc);
      const float bb = b1[nt * 16 + lo];
#pragma unroll
      for (int r = 0; r < 8; ++r)
        hp[(r + 8 * hi) * 128 + nt * 16 + lo] = (_Float16)silu_f(acc[r] + bb);
    }
  }
  __syncthreads();

  if (act) {
    // GEMM2: w = h W2 + b2 -> global
#pragma unroll 1
    for (int nt = 0; nt < 2; ++nt) {
      v8f acc = {};
#pragma unroll
      for (int kt = 0; kt < 4; ++kt)
        acc = wmma_f16(a_frag_lds(hp, 128, kt * 32), b_frag(pack, 40 + kt * 2 + nt), acc);
      const float bb = b2[nt * 16 + lo];
#pragma unroll
      for (int r = 0; r < 8; ++r)
        wenv[(size_t)(e0 + r + 8 * hi) * 32 + nt * 16 + lo] = acc[r] + bb;
    }
    // per-head Q/K tiles + logits (each N-tile == one head since HD==16)
#pragma unroll 1
    for (int t = 0; t < 16; ++t) {
      v8f aq = {}, ak = {};
#pragma unroll
      for (int kt = 0; kt < 2; ++kt)
        aq = wmma_f16(a_frag_lds(qp, 64, kt * 32), b_frag(pack, 48 + kt * 16 + t), aq);
#pragma unroll
      for (int kt = 0; kt < 4; ++kt)
        ak = wmma_f16(a_frag_lds(a1, 160, kt * 32), b_frag(pack, 80 + kt * 16 + t), ak);
#pragma unroll
      for (int r = 0; r < 8; ++r) {
        float p = hsum16(aq[r] * ak[r] * 0.25f); // 1/sqrt(HD) = 1/4
        p = fminf(fmaxf(p, -5.0f), 5.0f);
        if (lo == 0) {
          const int e = e0 + r + 8 * hi;
          logits[(size_t)e * 16 + t] = p;
          atomicMax(&mx[(size_t)esrc[e] * 16 + t], fenc(p));
        }
      }
    }
  }
}

// ---------------------------------------------------------------------------
// K2: per (edge,m): ew = exp(l - mx[src]); accumulate num/den into node bufs.
// ---------------------------------------------------------------------------
__global__ void k2_scatter(const float* __restrict__ equiv, const float* __restrict__ logits,
                           const float* __restrict__ wenv, const unsigned* __restrict__ mx,
                           const int* __restrict__ esrc,
                           float* __restrict__ den, float* __restrict__ nums,
                           float* __restrict__ numv, long long EM) {
  const long long idx = (long long)blockIdx.x * blockDim.x + threadIdx.x;
  if (idx >= EM) return;
  const int e = (int)(idx >> 4);
  const int m = (int)(idx & 15);
  const int s = esrc[e];
  const float ew = __expf(logits[idx] - fdec(mx[(size_t)s * 16 + m]));
  const float ws = wenv[(size_t)e * 32 + m];
  const float wv = wenv[(size_t)e * 32 + 16 + m];
  const float xs = equiv[(size_t)e * 64 + m];
  atomicAdd(&den[(size_t)s * 16 + m], ew);
  atomicAdd(&nums[(size_t)s * 16 + m], xs * ws * ew);
#pragma unroll
  for (int i = 0; i < 3; ++i) {
    const float xv = equiv[(size_t)e * 64 + 16 + 3 * m + i];
    atomicAdd(&numv[((size_t)s * 16 + m) * 3 + i], xv * wv * ew);
  }
}

// ---------------------------------------------------------------------------
// K3: per node (16 lanes): ns = num/(den+eps), then SO(3) layernorm.
// ---------------------------------------------------------------------------
__global__ void k3_node(const float* __restrict__ den, const float* __restrict__ nums,
                        const float* __restrict__ numv,
                        float* __restrict__ nsA, float* __restrict__ nvA, int N) {
  const int tid = threadIdx.x;
  const int node = blockIdx.x * 8 + (tid >> 4);
  const int m = tid & 15;
  if (node >= N) return;
  const size_t o = (size_t)node * 16 + m;
  const float d = frcp(den[o] + EPSV);
  const float s = nums[o] * d;
  const float v0 = numv[o * 3 + 0] * d;
  const float v1 = numv[o * 3 + 1] * d;
  const float v2 = numv[o * 3 + 2] * d;
  const float rs = frsq(hsum16(s * s) * 0.0625f + EPSV);
  const float rv = frsq(hsum16(v0 * v0 + v1 * v1 + v2 * v2) * 0.0625f + EPSV);
  nsA[o] = s * rs;
  nvA[o * 3 + 0] = v0 * rv;
  nvA[o * 3 + 1] = v1 * rv;
  nvA[o * 3 + 2] = v2 * rv;
}

// ---------------------------------------------------------------------------
// K4: per 16-edge group (1 wave): tensor products + LNs -> proj GEMMs ->
//     gates, mixing, residual -> outputs.
// ---------------------------------------------------------------------------
__global__ void __launch_bounds__(128)
k4_output(const float* __restrict__ scal, const float* __restrict__ equiv,
          const float* __restrict__ cond, const float* __restrict__ nsA,
          const float* __restrict__ nvA, const int* __restrict__ esrc,
          const float* __restrict__ bp1, const float* __restrict__ bp2,
          const float* __restrict__ wmix0, const float* __restrict__ wmix1,
          const float* __restrict__ ruc, const _Float16* __restrict__ pack,
          float* __restrict__ outS, float* __restrict__ outV, int E) {
  __shared__ _Float16 shh[4 * (16 * 64 + 16 * 128)];
  __shared__ float shg[4 * (16 * 32)];
  const int wave = threadIdx.x >> 5;
  const int l = threadIdx.x & 31;
  const int lo = l & 15, hi = l >> 4;
  _Float16* ap = shh + wave * (16 * 64 + 16 * 128); // [16][64]  [s00n|s11n|cond]
  _Float16* hp = ap + 16 * 64;                      // [16][128] hidden
  float* gp = shg + wave * (16 * 32);               // [16][32]  gates g0|g1
  const int e0 = (blockIdx.x * 4 + wave) * 16;
  const bool act = (e0 < E);
  const float c = ruc[0];
  const float co = frsq(c * c + 1.0f);
  const float cn = c * co;

  if (act) { // phase A: build invariant A-panel (m = lo, edge = hi + 2j)
#pragma unroll 1
    for (int j = 0; j < 8; ++j) {
      const int el = hi + 2 * j;
      const int e = e0 + el;
      const int m = lo;
      const int s = esrc[e];
      const float xs  = equiv[(size_t)e * 64 + m];
      const float xv0 = equiv[(size_t)e * 64 + 16 + 3 * m + 0];
      const float xv1 = equiv[(size_t)e * 64 + 16 + 3 * m + 1];
      const float xv2 = equiv[(size_t)e * 64 + 16 + 3 * m + 2];
      const float ys  = nsA[(size_t)s * 16 + m];
      const float yv0 = nvA[((size_t)s * 16 + m) * 3 + 0];
      const float yv1 = nvA[((size_t)s * 16 + m) * 3 + 1];
      const float yv2 = nvA[((size_t)s * 16 + m) * 3 + 2];
      const float s00 = xs * ys;
      const float s11 = (xv0 * yv0 + xv1 * yv1 + xv2 * yv2) * INV_SQRT3;
      const float rs00 = frsq(hsum16(s00 * s00) * 0.0625f + EPSV);
      const float rs11 = frsq(hsum16(s11 * s11) * 0.0625f + EPSV);
      ap[el * 64 + m]      = (_Float16)(s00 * rs00);
      ap[el * 64 + 16 + m] = (_Float16)(s11 * rs11);
      ap[el * 64 + 32 + m] = (_Float16)cond[(size_t)e * 32 + m];
      ap[el * 64 + 48 + m] = (_Float16)cond[(size_t)e * 32 + 16 + m];
    }
  }
  __syncthreads();

  if (act) { // GEMM5: [16,64]x[64,128] -> hp
#pragma unroll 1
    for (int nt = 0; nt < 8; ++nt) {
      v8f acc = {};
#pragma unroll
      for (int kt = 0; kt < 2; ++kt)
        acc = wmma_f16(a_frag_lds(ap, 64, kt * 32), b_frag(pack, 144 + kt * 8 + nt), acc);
      const float bb = bp1[nt * 16 + lo];
#pragma unroll
      for (int r = 0; r < 8; ++r)
        hp[(r + 8 * hi) * 128 + nt * 16 + lo] = (_Float16)silu_f(acc[r] + bb);
    }
  }
  __syncthreads();

  if (act) { // GEMM6: [16,128]x[128,160]; first 128 cols -> scalar out, last 32 -> gates
#pragma unroll 1
    for (int nt = 0; nt < 10; ++nt) {
      v8f acc = {};
#pragma unroll
      for (int kt = 0; kt < 4; ++kt)
        acc = wmma_f16(a_frag_lds(hp, 128, kt * 32), b_frag(pack, 160 + kt * 10 + nt), acc);
      const float bb = bp2[nt * 16 + lo];
#pragma unroll
      for (int r = 0; r < 8; ++r) {
        const float val = acc[r] + bb;
        const int e = e0 + r + 8 * hi;
        if (nt < 8) {
          const size_t oo = (size_t)e * 128 + nt * 16 + lo;
          outS[oo] = co * scal[oo] + cn * val;
        } else {
          gp[(r + 8 * hi) * 32 + (nt - 8) * 16 + lo] = val;
        }
      }
    }
  }
  __syncthreads();

  if (act) { // phase C: recompute products + vector LNs, gate, mix, residual
#pragma unroll 1
    for (int j = 0; j < 8; ++j) {
      const int el = hi + 2 * j;
      const int e = e0 + el;
      const int m = lo;
      const int s = esrc[e];
      const float xs  = equiv[(size_t)e * 64 + m];
      const float xv0 = equiv[(size_t)e * 64 + 16 + 3 * m + 0];
      const float xv1 = equiv[(size_t)e * 64 + 16 + 3 * m + 1];
      const float xv2 = equiv[(size_t)e * 64 + 16 + 3 * m + 2];
      const float ys  = nsA[(size_t)s * 16 + m];
      const float yv0 = nvA[((size_t)s * 16 + m) * 3 + 0];
      const float yv1 = nvA[((size_t)s * 16 + m) * 3 + 1];
      const float yv2 = nvA[((size_t)s * 16 + m) * 3 + 2];
      const float s00 = xs * ys;
      const float s11 = (xv0 * yv0 + xv1 * yv1 + xv2 * yv2) * INV_SQRT3;
      const float rs00 = frsq(hsum16(s00 * s00) * 0.0625f + EPSV);
      const float rs11 = frsq(hsum16(s11 * s11) * 0.0625f + EPSV);
      const float a01_0 = xs * yv0, a01_1 = xs * yv1, a01_2 = xs * yv2;
      const float a10_0 = xv0 * ys, a10_1 = xv1 * ys, a10_2 = xv2 * ys;
      const float a11_0 = (xv1 * yv2 - xv2 * yv1) * INV_SQRT2;
      const float a11_1 = (xv2 * yv0 - xv0 * yv2) * INV_SQRT2;
      const float a11_2 = (xv0 * yv1 - xv1 * yv0) * INV_SQRT2;
      const float r01 = frsq(hsum16(a01_0*a01_0 + a01_1*a01_1 + a01_2*a01_2) * 0.0625f + EPSV);
      const float r10 = frsq(hsum16(a10_0*a10_0 + a10_1*a10_1 + a10_2*a10_2) * 0.0625f + EPSV);
      const float r11 = frsq(hsum16(a11_0*a11_0 + a11_1*a11_1 + a11_2*a11_2) * 0.0625f + EPSV);
      const float g0 = gp[el * 32 + m];
      const float g1 = gp[el * 32 + 16 + m];
      const float os = g0 * (wmix0[m] * (s00 * rs00) + wmix0[16 + m] * (s11 * rs11));
      const float w1a = wmix1[m], w1b = wmix1[16 + m], w1c = wmix1[32 + m];
      const float ov0 = g1 * (w1a * a01_0 * r01 + w1b * a10_0 * r10 + w1c * a11_0 * r11);
      const float ov1 = g1 * (w1a * a01_1 * r01 + w1b * a10_1 * r10 + w1c * a11_1 * r11);
      const float ov2 = g1 * (w1a * a01_2 * r01 + w1b * a10_2 * r10 + w1c * a11_2 * r11);
      const size_t eb = (size_t)e * 64;
      outV[eb + m] = co * equiv[eb + m] + cn * os;
      outV[eb + 16 + 3 * m + 0] = co * equiv[eb + 16 + 3 * m + 0] + cn * ov0;
      outV[eb + 16 + 3 * m + 1] = co * equiv[eb + 16 + 3 * m + 1] + cn * ov1;
      outV[eb + 16 + 3 * m + 2] = co * equiv[eb + 16 + 3 * m + 2] + cn * ov2;
    }
  }
}

// ---------------------------------------------------------------------------
extern "C" void kernel_launch(void* const* d_in, const int* in_sizes, int n_in,
                              void* d_out, int out_size, void* d_ws, size_t ws_size,
                              hipStream_t stream) {
  const float* node_attrs = (const float*)d_in[0];
  const float* scal       = (const float*)d_in[1];
  const float* equiv      = (const float*)d_in[2];
  const float* cond       = (const float*)d_in[3];
  const float* ruc        = (const float*)d_in[4];
  const int*   esrc       = (const int*)d_in[5];
  const float* W_env1     = (const float*)d_in[6];
  const float* b_env1     = (const float*)d_in[7];
  const float* W_env2     = (const float*)d_in[8];
  const float* b_env2     = (const float*)d_in[9];
  const float* Wq         = (const float*)d_in[10];
  const float* Wk         = (const float*)d_in[11];
  const float* W_proj1    = (const float*)d_in[12];
  const float* b_proj1    = (const float*)d_in[13];
  const float* W_proj2    = (const float*)d_in[14];
  const float* b_proj2    = (const float*)d_in[15];
  const float* Wmix0      = (const float*)d_in[16];
  const float* Wmix1      = (const float*)d_in[17];

  const int E = in_sizes[1] / 128;
  const int N = in_sizes[0] / 64;

  char* ws = (char*)d_ws;
  _Float16* pack = (_Float16*)ws;          // 200 tiles * 512 halves = 200 KB
  size_t off = 262144;                     // 256 KB aligned start for f32 arrays
  float* logits = (float*)(ws + off); off += (size_t)E * 16 * 4;
  float* wenv   = (float*)(ws + off); off += (size_t)E * 32 * 4;
  const size_t zstart = off;
  unsigned* mx  = (unsigned*)(ws + off); off += (size_t)N * 16 * 4;
  float* den    = (float*)(ws + off); off += (size_t)N * 16 * 4;
  float* nums   = (float*)(ws + off); off += (size_t)N * 16 * 4;
  float* numv   = (float*)(ws + off); off += (size_t)N * 48 * 4;
  float* nsA    = (float*)(ws + off); off += (size_t)N * 16 * 4;
  float* nvA    = (float*)(ws + off); off += (size_t)N * 48 * 4;

  // zero mx/den/num_s/num_v (mx==0 decodes below any clipped logit)
  hipMemsetAsync(ws + zstart, 0, (size_t)N * 96 * 4, stream);

  pack_weights<<<200, 32, 0, stream>>>(W_env1, W_env2, Wq, Wk, W_proj1, W_proj2, pack);

  const int G = (E + 15) / 16;
  k1_edge_gemm<<<(G + 3) / 4, 128, 0, stream>>>(scal, cond, node_attrs, esrc,
                                                b_env1, b_env2, pack, logits, wenv, mx, E);

  const long long EM = (long long)E * 16;
  k2_scatter<<<(int)((EM + 255) / 256), 256, 0, stream>>>(equiv, logits, wenv, mx, esrc,
                                                          den, nums, numv, EM);

  k3_node<<<(N + 7) / 8, 128, 0, stream>>>(den, nums, numv, nsA, nvA, N);

  float* outS = (float*)d_out;
  float* outV = outS + (size_t)E * 128;
  k4_output<<<(G + 3) / 4, 128, 0, stream>>>(scal, equiv, cond, nsA, nvA, esrc,
                                             b_proj1, b_proj2, Wmix0, Wmix1, ruc,
                                             pack, outS, outV, E);
}
